// FusedTensorProductOp3_46497315947104
// MI455X (gfx1250) — matrix operand order repaired and checked by reading, not compiled, not run on hardware
//
#include <hip/hip_runtime.h>

// ---------------------------------------------------------------------------
// FusedTensorProductOp3 for MI455X (gfx1250, wave32, WMMA + async LDS staging)
//
// out[b, s2, u] = sum_p [idx2[p]==s2] * coef[p] * x0[b, idx0[p], u] * x1[b, idx1[p], u]
//
// Per batch:  Out_b(64x128) = M(64x256) @ Prod_b(256x128)
//   M[s2,p]    = coef[p] * (idx2[p]==s2)          (bf16, built once per block)
//   Prod_b[p,u]= x0[b,idx0[p],u]*x1[b,idx1[p],u]  (bf16, rebuilt per batch in LDS)
// Accumulation in fp32 via v_wmma_f32_16x16x32_bf16.
// x0/x1 staged global->LDS via GLOBAL_LOAD_ASYNC_TO_LDS_B128 (ASYNCcnt path).
// ---------------------------------------------------------------------------

typedef __attribute__((ext_vector_type(16))) __bf16 v16bf;
typedef __attribute__((ext_vector_type(8)))  float  v8f;
typedef __attribute__((ext_vector_type(4)))  int    i32x4;

typedef __attribute__((address_space(1))) i32x4* gptr_i32x4;
typedef __attribute__((address_space(3))) i32x4* lptr_i32x4;

#define B_TOT 4096
#define S0D   64
#define S1D   16
#define S2D   64
#define UD    128
#define PD    256

// LDS layout (bytes), dynamic shared:
//   sh_x0   : float [64*128]        @ 0       (32768)
//   sh_x1   : float [16*128]        @ 32768   ( 8192)
//   sh_meta : int2  [256]           @ 40960   ( 2048)
//   sh_m    : u32   [64*129]        @ 43008   (33024)  M as bf16 pairs, row stride 129 dw
//   sh_prod : u32   [128*129]       @ 76032   (66048)  Prod^T [u][p] bf16 pairs, stride 129 dw
#define SMEM_BYTES 142080

#if __has_builtin(__builtin_amdgcn_global_load_async_to_lds_b128)
#define HAVE_ASYNC_LDS 1
#else
#define HAVE_ASYNC_LDS 0
#endif

union Frag16 { unsigned u[8]; v16bf v; };

__device__ __forceinline__ unsigned pack2_bf16(float a, float b) {
  unsigned ua = __float_as_uint(a);
  unsigned ub = __float_as_uint(b);
  ua += 0x7FFFu + ((ua >> 16) & 1u);   // round-to-nearest-even
  ub += 0x7FFFu + ((ub >> 16) & 1u);
  return (ub & 0xFFFF0000u) | (ua >> 16);
}

__device__ __forceinline__ unsigned short f2bf(float a) {
  unsigned ua = __float_as_uint(a);
  ua += 0x7FFFu + ((ua >> 16) & 1u);
  return (unsigned short)(ua >> 16);
}

__device__ __forceinline__ void wait_async_zero() {
#if __has_builtin(__builtin_amdgcn_s_wait_asynccnt)
  __builtin_amdgcn_s_wait_asynccnt(0);
#else
  asm volatile("s_wait_asynccnt 0x0" ::: "memory");
#endif
}

__global__ __launch_bounds__(256)
void ftp3_wmma_kernel(const float* __restrict__ x0, const float* __restrict__ x1,
                      const float* __restrict__ coef, const int* __restrict__ idx0,
                      const int* __restrict__ idx1, const int* __restrict__ idx2,
                      float* __restrict__ out)
{
  extern __shared__ char smem[];
  float*    sh_x0   = (float*)(smem);
  float*    sh_x1   = (float*)(smem + 32768);
  int2*     sh_meta = (int2*)(smem + 40960);
  unsigned* sh_m    = (unsigned*)(smem + 43008);
  unsigned* sh_prod = (unsigned*)(smem + 76032);

  const int t    = threadIdx.x;
  const int lane = t & 31;
  const int w    = t >> 5;          // 8 waves

  // ---- one-time per block: zero M, then scatter coef into M; stage index meta
  for (int i = t; i < 64 * 129; i += 256) sh_m[i] = 0u;
  __syncthreads();
  if (t < PD) {
    sh_meta[t] = make_int2(idx0[t] * UD, idx1[t] * UD);
    ((unsigned short*)sh_m)[idx2[t] * 258 + t] = f2bf(coef[t]);
  }

  // ---- per-wave GEMM tile coordinates: 2 s2-tiles x 2 u-tiles per wave
  const int sbase   = (w >> 2) * 2;           // s2-tile base: {0,2}
  const int ubase   = (w & 3) * 2;            // u-tile base : {0,2,4,6}
  const int n16     = lane & 15;
  const int aColOff = (lane < 16) ? 0 : 8;    // A: bf16 K-base 0 / 16 -> dw 0 / 8
  const int bColOff = (lane < 16) ? 0 : 8;    // B: bf16 K-base 0 / 16 -> dw 0 / 8
  const int mRow0   = (sbase * 16 + n16) * 129;
  const int mRow1   = mRow0 + 16 * 129;
  const int pRow0   = (ubase * 16 + n16) * 129;
  const int pRow1   = pRow0 + 16 * 129;
  const int mbase   = (lane < 16) ? 0 : 8;    // C/D row offset

  // ---- prod-phase coordinates: thread handles 2 u-rows x 64 p (in pairs)
  const int u2 = (t & 63) * 2;
  const int pg = t >> 6;

  for (int b = blockIdx.x; b < B_TOT; b += gridDim.x) {
    const float* x0b  = x0  + (size_t)b * (S0D * UD);
    const float* x1b  = x1  + (size_t)b * (S1D * UD);
    float*       outb = out + (size_t)b * (S2D * UD);

    // ---- stage x0/x1 rows for this batch into LDS
#if HAVE_ASYNC_LDS
    {
      const float4* g0 = (const float4*)x0b;
      float4*       s0 = (float4*)sh_x0;
      #pragma unroll
      for (int r = 0; r < (S0D * UD) / 4 / 256; ++r) {   // 8 iters
        const int i = t + r * 256;
        __builtin_amdgcn_global_load_async_to_lds_b128(
            (gptr_i32x4)(g0 + i), (lptr_i32x4)(s0 + i), 0, 0);
      }
      const float4* g1 = (const float4*)x1b;
      float4*       s1 = (float4*)sh_x1;
      #pragma unroll
      for (int r = 0; r < (S1D * UD) / 4 / 256; ++r) {   // 2 iters
        const int i = t + r * 256;
        __builtin_amdgcn_global_load_async_to_lds_b128(
            (gptr_i32x4)(g1 + i), (lptr_i32x4)(s1 + i), 0, 0);
      }
      wait_async_zero();
    }
#else
    {
      const float4* g0 = (const float4*)x0b;
      float4*       s0 = (float4*)sh_x0;
      for (int i = t; i < (S0D * UD) / 4; i += 256) s0[i] = g0[i];
      const float4* g1 = (const float4*)x1b;
      float4*       s1 = (float4*)sh_x1;
      for (int i = t; i < (S1D * UD) / 4; i += 256) s1[i] = g1[i];
    }
#endif
    __syncthreads();  // staging visible; also fences previous iteration's GEMM reads

    // ---- build Prod^T[u][p] in bf16 (pairs of p packed per dword)
    #pragma unroll 4
    for (int i = 0; i < 32; ++i) {
      const int p = pg * 64 + 2 * i;
      const int2 ma = sh_meta[p];
      const int2 mb = sh_meta[p + 1];
      const float2 a0 = *(const float2*)(sh_x0 + ma.x + u2);
      const float2 a1 = *(const float2*)(sh_x1 + ma.y + u2);
      const float2 b0 = *(const float2*)(sh_x0 + mb.x + u2);
      const float2 b1 = *(const float2*)(sh_x1 + mb.y + u2);
      const int pc = pg * 32 + i;
      sh_prod[u2 * 129 + pc]       = pack2_bf16(a0.x * a1.x, b0.x * b1.x);
      sh_prod[(u2 + 1) * 129 + pc] = pack2_bf16(a0.y * a1.y, b0.y * b1.y);
    }
    __syncthreads();

    // ---- GEMM: Out_b = M @ Prod_b, K=256 in 8 steps of 32, 2x2 tiles per wave
    v8f acc00 = {}, acc01 = {}, acc10 = {}, acc11 = {};
    #pragma unroll
    for (int kk = 0; kk < 8; ++kk) {
      const int kc = kk * 16;
      Frag16 A0, A1, Bf0, Bf1;
      #pragma unroll
      for (int i = 0; i < 4; ++i) {
        A0.u[i]     = sh_m[mRow0 + kc + aColOff + i];
        A0.u[4 + i] = sh_m[mRow0 + kc + 4 + aColOff + i];
        A1.u[i]     = sh_m[mRow1 + kc + aColOff + i];
        A1.u[4 + i] = sh_m[mRow1 + kc + 4 + aColOff + i];
      }
      #pragma unroll
      for (int j = 0; j < 8; ++j) {
        Bf0.u[j] = sh_prod[pRow0 + kc + bColOff + j];
        Bf1.u[j] = sh_prod[pRow1 + kc + bColOff + j];
      }
      acc00 = __builtin_amdgcn_wmma_f32_16x16x32_bf16(false, A0.v, false, Bf0.v,
                                                      (short)0, acc00, false, false);
      acc01 = __builtin_amdgcn_wmma_f32_16x16x32_bf16(false, A0.v, false, Bf1.v,
                                                      (short)0, acc01, false, false);
      acc10 = __builtin_amdgcn_wmma_f32_16x16x32_bf16(false, A1.v, false, Bf0.v,
                                                      (short)0, acc10, false, false);
      acc11 = __builtin_amdgcn_wmma_f32_16x16x32_bf16(false, A1.v, false, Bf1.v,
                                                      (short)0, acc11, false, false);
    }

    // ---- store accumulators (every output element written exactly once)
    #pragma unroll
    for (int r = 0; r < 8; ++r) {
      const int row0 = (sbase * 16 + mbase + r) * UD;
      const int row1 = ((sbase + 1) * 16 + mbase + r) * UD;
      outb[row0 + ubase * 16 + n16]       = acc00[r];
      outb[row0 + (ubase + 1) * 16 + n16] = acc01[r];
      outb[row1 + ubase * 16 + n16]       = acc10[r];
      outb[row1 + (ubase + 1) * 16 + n16] = acc11[r];
    }
    __syncthreads();  // protect sh_prod/sh_x0 overwrite on next iteration
  }
}

extern "C" void kernel_launch(void* const* d_in, const int* in_sizes, int n_in,
                              void* d_out, int out_size, void* d_ws, size_t ws_size,
                              hipStream_t stream) {
  (void)in_sizes; (void)n_in; (void)out_size; (void)d_ws; (void)ws_size;
  const float* x0   = (const float*)d_in[0];
  const float* x1   = (const float*)d_in[1];
  const float* coef = (const float*)d_in[2];
  const int*   idx0 = (const int*)d_in[3];
  const int*   idx1 = (const int*)d_in[4];
  const int*   idx2 = (const int*)d_in[5];
  float* out = (float*)d_out;

  dim3 grid(1024);   // 4 batches per block via grid-stride
  dim3 block(256);   // 8 wave32
  ftp3_wmma_kernel<<<grid, block, SMEM_BYTES, stream>>>(x0, x1, coef, idx0, idx1, idx2, out);
}
